// Attention_32839319945607
// MI455X (gfx1250) — compile-verified, matrix operand order checked by reading, not run on hardware
//
#include <hip/hip_runtime.h>
#include <hip/hip_bf16.h>

typedef unsigned short u16;
typedef __attribute__((ext_vector_type(16))) __bf16 v16bf;
typedef __attribute__((ext_vector_type(8)))  float  v8f;
typedef __attribute__((ext_vector_type(16))) unsigned short v16u;
typedef __attribute__((ext_vector_type(8)))  unsigned short u16x8;

union AB {
    v16bf bf;
    v16u  u;
    u16x8 h[2];
};

// Native converts: v_cvt_pk_bf16_f32 on gfx1250.
__device__ __forceinline__ __bf16 cvtbf(float f) { return (__bf16)f; }
__device__ __forceinline__ u16 cvtbf_bits(float f) {
    return __builtin_bit_cast(u16, (__bf16)f);
}

__device__ __forceinline__ v8f wmma_bf16(v16bf a, v16bf b, v8f c) {
    return __builtin_amdgcn_wmma_f32_16x16x32_bf16(false, a, false, b, (short)0, c, false, false);
}

__device__ __forceinline__ float hmax16(float v) {
    v = fmaxf(v, __shfl_xor(v, 1, 32));
    v = fmaxf(v, __shfl_xor(v, 2, 32));
    v = fmaxf(v, __shfl_xor(v, 4, 32));
    v = fmaxf(v, __shfl_xor(v, 8, 32));
    return v;
}
__device__ __forceinline__ float hsum16(float v) {
    v += __shfl_xor(v, 1, 32);
    v += __shfl_xor(v, 2, 32);
    v += __shfl_xor(v, 4, 32);
    v += __shfl_xor(v, 8, 32);
    return v;
}

// Async global -> LDS copy, 128b per lane.  VDST = LDS byte offset (low 32
// bits of the generic pointer, per the CDNA5 LDS aperture rules), VADDR = the
// 64-bit global address.  Tracked by ASYNCcnt.
__device__ __forceinline__ void async_b128(const u16* g, u16* lds) {
    unsigned loff = (unsigned)(uintptr_t)(void*)lds;
    unsigned long long ga = (unsigned long long)(uintptr_t)g;
    asm volatile("global_load_async_to_lds_b128 %0, %1, off"
                 :: "v"(loff), "v"(ga) : "memory");
}
__device__ __forceinline__ void wait_async0() {
    asm volatile("s_wait_asynccnt 0x0" ::: "memory");
}

// ---------------------------------------------------------------------------
#define BATCH   8
#define SEQ     2048
#define DMODEL  384
#define HEADS   8
#define DK      48
#define DKP     64
#define LSTR    72      // padded LDS row stride (u16): 144B = 36 banks, no conflicts
#define MROWS   (BATCH * SEQ)
#define E_QKV   (3 * DMODEL)

// ---------------------------------------------------------------------------
__global__ void prep_weights(const float* __restrict__ Wqkv,
                             const float* __restrict__ Wo,
                             u16* __restrict__ Wqkvt,   // [1152][384]
                             u16* __restrict__ Wot) {   // [384][384]
    int i = blockIdx.x * blockDim.x + threadIdx.x;
    const int n1 = E_QKV * DMODEL;
    if (i < n1) {
        int e = i / DMODEL, d = i - e * DMODEL;
        Wqkvt[i] = cvtbf_bits(Wqkv[d * E_QKV + e]);
    }
    int j = i - n1;
    if (j >= 0 && j < DMODEL * DMODEL) {
        int e = j / DMODEL, d = j - e * DMODEL;
        Wot[j] = cvtbf_bits(Wo[d * DMODEL + e]);
    }
}

__global__ void zero_pad(u16* __restrict__ Qp, u16* __restrict__ Kp) {
    int row = blockIdx.x * blockDim.x + threadIdx.x;
    if (row < HEADS * BATCH * SEQ) {
        uint4 z = {0u, 0u, 0u, 0u};
        *(uint4*)(Qp + (size_t)row * DKP + 48) = z;
        *(uint4*)(Qp + (size_t)row * DKP + 56) = z;
        *(uint4*)(Kp + (size_t)row * DKP + 48) = z;
        *(uint4*)(Kp + (size_t)row * DKP + 56) = z;
    }
}

// ---------------------------------------------------------------------------
// Kernel 1: QKV projection -> head-major bf16 Qp/Kp (padded) and transposed Vt
// ---------------------------------------------------------------------------
__global__ void qkv_gemm(const float* __restrict__ x,
                         const u16* __restrict__ Wqkvt,
                         u16* __restrict__ Qp, u16* __restrict__ Kp,
                         u16* __restrict__ Vt) {
    const int wave = threadIdx.x >> 5, lane = threadIdx.x & 31;
    const int l16 = lane & 15, half = lane >> 4;
    const int rowbase = blockIdx.x * 128 + wave * 16;
    const int colbase = blockIdx.y * 64;

    v8f acc[4] = {v8f{}, v8f{}, v8f{}, v8f{}};
    const float* xrow = x + (size_t)(rowbase + l16) * DMODEL;

    for (int k = 0; k < DMODEL; k += 32) {
        AB a;
        const float* pa0 = xrow + k + 8 * half;
        const float* pa1 = xrow + k + 16 + 8 * half;
        float4 fa = *(const float4*)(pa0);
        float4 fb = *(const float4*)(pa0 + 4);
        float4 fc = *(const float4*)(pa1);
        float4 fd = *(const float4*)(pa1 + 4);
        a.bf[0] = cvtbf(fa.x); a.bf[1] = cvtbf(fa.y); a.bf[2] = cvtbf(fa.z); a.bf[3] = cvtbf(fa.w);
        a.bf[4] = cvtbf(fb.x); a.bf[5] = cvtbf(fb.y); a.bf[6] = cvtbf(fb.z); a.bf[7] = cvtbf(fb.w);
        a.bf[8]  = cvtbf(fc.x); a.bf[9]  = cvtbf(fc.y); a.bf[10] = cvtbf(fc.z); a.bf[11] = cvtbf(fc.w);
        a.bf[12] = cvtbf(fd.x); a.bf[13] = cvtbf(fd.y); a.bf[14] = cvtbf(fd.z); a.bf[15] = cvtbf(fd.w);

#pragma unroll
        for (int nt = 0; nt < 4; nt++) {
            const u16* wrow = Wqkvt + (size_t)(colbase + nt * 16 + l16) * DMODEL;
            AB b;
            b.h[0] = *(const u16x8*)(wrow + k + 8 * half);
            b.h[1] = *(const u16x8*)(wrow + k + 16 + 8 * half);
            acc[nt] = wmma_bf16(a.bf, b.bf, acc[nt]);
        }
    }

    const int b  = rowbase >> 11;
    const int n0 = (rowbase & 2047) + 8 * half;
#pragma unroll
    for (int nt = 0; nt < 4; nt++) {
        const int e   = colbase + nt * 16 + l16;
        const int seg = e / DMODEL;
        const int e2  = e - seg * DMODEL;
        const int hh  = e2 / DK, dd = e2 - hh * DK;
        if (seg < 2) {
            u16* base = (seg == 0 ? Qp : Kp) +
                        ((size_t)(b * HEADS + hh) * SEQ + n0) * DKP + dd;
#pragma unroll
            for (int r = 0; r < 8; r++)
                base[(size_t)r * DKP] = cvtbf_bits(acc[nt][r]);
        } else {
            u16* base = Vt + ((size_t)(b * HEADS + hh) * DK + dd) * SEQ + n0;
#pragma unroll
            for (int r = 0; r < 8; r++)
                base[r] = cvtbf_bits(acc[nt][r]);
        }
    }
}

// ---------------------------------------------------------------------------
// Kernel 2: flash attention with async double-buffered K/V staging.
// grid (bh=64, rowblk=16); 8 waves x 16 rows; 64-key chunks, 14 WMMAs/chunk.
// ---------------------------------------------------------------------------
__global__ void flash_attn(const u16* __restrict__ Qp, const u16* __restrict__ Kp,
                           const u16* __restrict__ Vt, u16* __restrict__ AO) {
    __shared__ u16 Kb[2][64][LSTR];   // 18 KB : K chunk, double buffered
    __shared__ u16 Vb[2][48][LSTR];   // 13.5 KB : V chunk (feature-major)
    __shared__ u16 Pl[8][16][LSTR];   // 18 KB : per-wave P tiles

    const int wave = threadIdx.x >> 5, lane = threadIdx.x & 31;
    const int l16 = lane & 15, half = lane >> 4;
    const int bh = blockIdx.x;
    const int rowbase = blockIdx.y * 128 + wave * 16;
    const int b = bh >> 3, hh = bh & 7;

    const u16* Q  = Qp + (size_t)bh * SEQ * DKP;
    const u16* Kg = Kp + (size_t)bh * SEQ * DKP;
    const u16* Vg = Vt + (size_t)bh * DK * SEQ;

    // Cooperative async stage of one 64-key chunk:
    //   K: 64 rows x 128B = 512 x 16B segs;  V: 48 rows x 128B = 384 segs.
    //   896 segs = 28 lane-wide b128 instructions spread over 8 waves.
    auto stage = [&](int j, int buf) {
#pragma unroll
        for (int i = 0; i < 4; i++) {
            int inst = wave + 8 * i;
            if (inst < 28) {
                int seg = inst * 32 + lane;
                if (seg < 512) {
                    int row = seg >> 3, s8 = (seg & 7) * 8;
                    async_b128(Kg + (size_t)(j + row) * DKP + s8,
                               &Kb[buf][row][s8]);
                } else {
                    int s2 = seg - 512;
                    int row = s2 >> 3, s8 = (s2 & 7) * 8;
                    async_b128(Vg + (size_t)row * SEQ + j + s8,
                               &Vb[buf][row][s8]);
                }
            }
        }
    };

    // Q A-fragments (direct global, once per wave)
    const u16* qrow = Q + (size_t)(rowbase + l16) * DKP;
    AB qa0, qa1;
    qa0.h[0] = *(const u16x8*)(qrow + 8 * half);
    qa0.h[1] = *(const u16x8*)(qrow + 16 + 8 * half);
    qa1.h[0] = *(const u16x8*)(qrow + 32 + 8 * half);
    qa1.h[1] = *(const u16x8*)(qrow + 48 + 8 * half);

    v8f o0 = {}, o1 = {}, o2 = {};
    float mi[8], li[8];
#pragma unroll
    for (int r = 0; r < 8; r++) { mi[r] = -3.0e38f; li[r] = 0.0f; }

    const float scale = 0.022097086912079608f;   // 1/sqrt(2048)
    u16* pw = &Pl[wave][0][0];

    stage(0, 0);
    wait_async0();
    __syncthreads();

    for (int j = 0; j < SEQ; j += 64) {
        const int cur = (j >> 6) & 1;
        if (j + 64 < SEQ) stage(j + 64, cur ^ 1);   // overlap with compute

        // S tiles t=0..3 from LDS K
        v8f s[4];
#pragma unroll
        for (int t = 0; t < 4; t++) {
            const u16* kr = &Kb[cur][16 * t + l16][0];
            AB k0, k1;
            k0.h[0] = *(const u16x8*)(kr + 8 * half);
            k0.h[1] = *(const u16x8*)(kr + 16 + 8 * half);
            k1.h[0] = *(const u16x8*)(kr + 32 + 8 * half);
            k1.h[1] = *(const u16x8*)(kr + 48 + 8 * half);
            v8f c = {};
            c = wmma_bf16(qa0.bf, k0.bf, c);
            c = wmma_bf16(qa1.bf, k1.bf, c);
            s[t] = c;
        }

        // online softmax across 64 columns; write P straight into LDS
#pragma unroll
        for (int r = 0; r < 8; r++) {
            float a0 = s[0][r] * scale, a1 = s[1][r] * scale;
            float a2 = s[2][r] * scale, a3 = s[3][r] * scale;
            float mx = hmax16(fmaxf(fmaxf(a0, a1), fmaxf(a2, a3)));
            float mn = fmaxf(mi[r], mx);
            float alpha = __expf(mi[r] - mn);
            mi[r] = mn;
            float p0 = __expf(a0 - mn), p1 = __expf(a1 - mn);
            float p2 = __expf(a2 - mn), p3 = __expf(a3 - mn);
            u16* pr = pw + (r + 8 * half) * LSTR + l16;
            pr[0]  = cvtbf_bits(p0);
            pr[16] = cvtbf_bits(p1);
            pr[32] = cvtbf_bits(p2);
            pr[48] = cvtbf_bits(p3);
            float rs = hsum16((p0 + p1) + (p2 + p3));
            li[r] = li[r] * alpha + rs;
            o0[r] *= alpha; o1[r] *= alpha; o2[r] *= alpha;
        }

        // P back as two A fragments (per-wave buffer; DS ops are in-order)
        const u16* prow = pw + l16 * LSTR;
        AB pa0, pa1;
        pa0.h[0] = *(const u16x8*)(prow + 8 * half);
        pa0.h[1] = *(const u16x8*)(prow + 16 + 8 * half);
        pa1.h[0] = *(const u16x8*)(prow + 32 + 8 * half);
        pa1.h[1] = *(const u16x8*)(prow + 48 + 8 * half);

        // O += P(16x64) x V(64x48) from LDS V
#pragma unroll
        for (int ft = 0; ft < 3; ft++) {
            const u16* vr = &Vb[cur][16 * ft + l16][0];
            AB v0, v1;
            v0.h[0] = *(const u16x8*)(vr + 8 * half);
            v0.h[1] = *(const u16x8*)(vr + 16 + 8 * half);
            v1.h[0] = *(const u16x8*)(vr + 32 + 8 * half);
            v1.h[1] = *(const u16x8*)(vr + 48 + 8 * half);
            v8f* o = (ft == 0) ? &o0 : (ft == 1) ? &o1 : &o2;
            *o = wmma_bf16(pa0.bf, v0.bf, *o);
            *o = wmma_bf16(pa1.bf, v1.bf, *o);
        }

        wait_async0();      // this wave's share of the next chunk landed
        __syncthreads();    // whole block: next chunk ready, cur fully read
    }

    // normalize + store bf16 into AO[16384][384]
#pragma unroll
    for (int r = 0; r < 8; r++) {
        float inv = 1.0f / li[r];
        int gm = b * SEQ + rowbase + r + 8 * half;
        u16* dst = AO + (size_t)gm * DMODEL + hh * DK + l16;
        dst[0]  = cvtbf_bits(o0[r] * inv);
        dst[16] = cvtbf_bits(o1[r] * inv);
        dst[32] = cvtbf_bits(o2[r] * inv);
    }
}

// ---------------------------------------------------------------------------
// Kernel 3: output projection.  AO bf16 x Wot + b_o -> out fp32
// ---------------------------------------------------------------------------
__global__ void out_gemm(const u16* __restrict__ AO, const u16* __restrict__ Wot,
                         const float* __restrict__ bo, float* __restrict__ out) {
    const int wave = threadIdx.x >> 5, lane = threadIdx.x & 31;
    const int l16 = lane & 15, half = lane >> 4;
    const int rowbase = blockIdx.x * 128 + wave * 16;
    const int colbase = blockIdx.y * 64;

    v8f acc[4] = {v8f{}, v8f{}, v8f{}, v8f{}};
    const u16* arow = AO + (size_t)(rowbase + l16) * DMODEL;

    for (int k = 0; k < DMODEL; k += 32) {
        AB a;
        a.h[0] = *(const u16x8*)(arow + k + 8 * half);
        a.h[1] = *(const u16x8*)(arow + k + 16 + 8 * half);
#pragma unroll
        for (int nt = 0; nt < 4; nt++) {
            const u16* wrow = Wot + (size_t)(colbase + nt * 16 + l16) * DMODEL;
            AB bfr;
            bfr.h[0] = *(const u16x8*)(wrow + k + 8 * half);
            bfr.h[1] = *(const u16x8*)(wrow + k + 16 + 8 * half);
            acc[nt] = wmma_bf16(a.bf, bfr.bf, acc[nt]);
        }
    }

#pragma unroll
    for (int nt = 0; nt < 4; nt++) {
        const int col = colbase + nt * 16 + l16;
        const float bias = bo[col];
        float* dst = out + (size_t)(rowbase + 8 * half) * DMODEL + col;
#pragma unroll
        for (int r = 0; r < 8; r++)
            dst[(size_t)r * DMODEL] = acc[nt][r] + bias;
    }
}

// ---------------------------------------------------------------------------
extern "C" void kernel_launch(void* const* d_in, const int* in_sizes, int n_in,
                              void* d_out, int out_size, void* d_ws, size_t ws_size,
                              hipStream_t stream) {
    const float* x    = (const float*)d_in[0];
    const float* Wqkv = (const float*)d_in[1];
    const float* Wo   = (const float*)d_in[2];
    const float* bo   = (const float*)d_in[3];
    float* out = (float*)d_out;

    char* ws = (char*)d_ws;
    const size_t SZ_WQKVT = (size_t)E_QKV * DMODEL * 2;
    const size_t SZ_WOT   = (size_t)DMODEL * DMODEL * 2;
    const size_t SZ_QP    = (size_t)BATCH * HEADS * SEQ * DKP * 2;
    const size_t SZ_VT    = (size_t)BATCH * HEADS * DK * SEQ * 2;

    u16* Wqkvt = (u16*)(ws);
    u16* Wot   = (u16*)(ws + SZ_WQKVT);
    u16* Qp    = (u16*)(ws + SZ_WQKVT + SZ_WOT);
    u16* Kp    = (u16*)(ws + SZ_WQKVT + SZ_WOT + SZ_QP);
    u16* Vt    = (u16*)(ws + SZ_WQKVT + SZ_WOT + 2 * SZ_QP);
    u16* AO    = (u16*)(ws + SZ_WQKVT + SZ_WOT + 2 * SZ_QP + SZ_VT);
    (void)in_sizes; (void)n_in; (void)out_size; (void)ws_size;

    {
        int total = E_QKV * DMODEL + DMODEL * DMODEL;
        prep_weights<<<(total + 255) / 256, 256, 0, stream>>>(Wqkv, Wo, Wqkvt, Wot);
        int rows = BATCH * HEADS * SEQ;
        zero_pad<<<(rows + 255) / 256, 256, 0, stream>>>(Qp, Kp);
    }
    qkv_gemm<<<dim3(MROWS / 128, E_QKV / 64), 256, 0, stream>>>(x, Wqkvt, Qp, Kp, Vt);
    flash_attn<<<dim3(BATCH * HEADS, SEQ / 128), 256, 0, stream>>>(Qp, Kp, Vt, AO);
    out_gemm<<<dim3(MROWS / 128, DMODEL / 64), 256, 0, stream>>>(AO, Wot, bo, out);
}
